// SingleNodeReadout_82678120448502
// MI455X (gfx1250) — compile-verified, matrix operand order checked by reading, not run on hardware
//
#include <hip/hip_runtime.h>
#include <hip/hip_bf16.h>

typedef __attribute__((ext_vector_type(16))) _Float16 v16h;
typedef __attribute__((ext_vector_type(8)))  float    v8f;

#define BM 128
#define BN 128
#define BK 32
#define LDSS 40   // LDS row stride in halves (80 B) -> conflict-free ds_load_b128 fragment reads

// ---------------- prep kernels ----------------

__global__ void k_zero_f32(float* __restrict__ p, int nElem) {
  int i = blockIdx.x * blockDim.x + threadIdx.x;
  if (i < nElem) p[i] = 0.f;
}

// flatT[c][k] (c in [0,768), k in [0,128)) = flat[k][c], flat = mixer_x 'B t p f -> (p B) t f'
__global__ void k_flatT(const float* __restrict__ mixer, _Float16* __restrict__ flatT) {
  int idx = blockIdx.x * blockDim.x + threadIdx.x;
  if (idx >= 768 * 128) return;
  int k = idx & 127, c = idx >> 7;
  int b = k & 3, pi = k >> 2;        // k = pi*B + b, B = 4
  int ti = c >> 6, fi = c & 63;      // c = ti*f + fi, f = 64
  flatT[idx] = (_Float16)mixer[(((b * 12 + ti) * 32) + pi) * 64 + fi];
}

// histogram of (output-row, source-row) membership; segment id (node*B+b) -> row (b*n+node)
__global__ void k_hist(const int* __restrict__ sb, const int* __restrict__ mapper,
                       float* __restrict__ H, int E, int n) {
  int e = blockIdx.x * blockDim.x + threadIdx.x;
  if (e >= E) return;
  int seg = mapper[e];
  int b = seg & 3, node = seg >> 2;  // B = 4
  atomicAdd(&H[(size_t)(b * n + node) * 128 + sb[e]], 1.0f);
}

// convert H to f16 and produce per-row 1/max(count,1)
__global__ __launch_bounds__(128)
void k_finish_h(const float* __restrict__ H, _Float16* __restrict__ Hf,
                float* __restrict__ inv, int Mreal) {
  __shared__ float red[128];
  int row = blockIdx.x, tid = threadIdx.x;
  float v = H[(size_t)row * 128 + tid];
  Hf[(size_t)row * 128 + tid] = (_Float16)v;
  red[tid] = v;
  __syncthreads();
  for (int s = 64; s > 0; s >>= 1) {
    if (tid < s) red[tid] += red[tid + s];
    __syncthreads();
  }
  if (tid == 0) {
    float c = red[0];
    inv[row] = (row < Mreal) ? (1.f / (c < 1.f ? 1.f : c)) : 0.f;
  }
}

// transpose + pad weights to f16: Wt[nd][kd], zero outside (Nsrc, Ksrc)
__global__ void k_prep_w(const float* __restrict__ W, _Float16* __restrict__ Wt,
                         int Nsrc, int Ksrc, int Npad, int Kpad) {
  int idx = blockIdx.x * blockDim.x + threadIdx.x;
  if (idx >= Npad * Kpad) return;
  int kd = idx % Kpad, nd = idx / Kpad;
  float v = (nd < Nsrc && kd < Ksrc) ? W[(size_t)kd * Nsrc + nd] : 0.f;
  Wt[idx] = (_Float16)v;
}

// fill passthrough cols [0,12) and K-pad cols [780,800) of the MLP input
__global__ void k_edges(const float* __restrict__ features, _Float16* __restrict__ act,
                        int Mreal) {
  int row = blockIdx.x, tid = threadIdx.x;  // 32 threads
  if (tid < 12) {
    float v = (row < Mreal) ? features[(size_t)row * 12 + tid] : 0.f;
    act[(size_t)row * 800 + tid] = (_Float16)v;
  } else {
    act[(size_t)row * 800 + 768 + tid] = (_Float16)0.f;  // cols 780..799
  }
}

// ---------------- async copy helpers (CDNA5 async global->LDS, ASYNCcnt) ----------------

__device__ __forceinline__ void async_b128(const _Float16* g, const _Float16* l) {
  // low 32 bits of a generic LDS pointer are the wave-relative LDS byte address
  unsigned lds = (unsigned)(size_t)l;
  asm volatile("global_load_async_to_lds_b128 %0, %1, off" :: "v"(lds), "v"(g) : "memory");
}

// ---------------- WMMA GEMM ----------------
// out = A(MxK,f16, stride K) @ Wt^T (Wt is [Npad][K], f16)
// MODE 0: outH = relu(out + bias) f16, zero-fill cols [n_valid,out_stride)
// MODE 1: outF = out + bias (f32) guarded to (m_valid, n_valid)
// MODE 2: outH[:, 12+col] = out * scale[row] (nodes-mean epilogue)
template <int MODE>
__global__ __launch_bounds__(256)
void k_gemm(const _Float16* __restrict__ A, const _Float16* __restrict__ Wt,
            const float* __restrict__ bias, const float* __restrict__ scale,
            _Float16* __restrict__ outH, float* __restrict__ outF,
            int K, int n_valid, int out_stride, int m_valid) {
  __shared__ _Float16 As[2][BM * LDSS];
  __shared__ _Float16 Bs[2][BN * LDSS];

  const int tid = threadIdx.x;
  const int lane = tid & 31;
  const int wid = tid >> 5;               // 8 waves
  const int wm = wid >> 1, wn = wid & 1;  // 4 x 2 wave grid; wave tile 32 x 64
  const int rowBlock = blockIdx.x * BM;
  const int colBlock = blockIdx.y * BN;

  const int r0 = tid >> 2;                // 0..63
  const int off0 = (tid & 3) * 8;         // 16B chunk within a 32-half row

  v8f zero = {};
  v8f acc[2][4];
#pragma unroll
  for (int i = 0; i < 2; ++i)
#pragma unroll
    for (int j = 0; j < 4; ++j) acc[i][j] = zero;

  const int lrow  = lane & 15;
  const int lhalf = (lane >> 4) * 8;      // K-half select per ISA A/B fragment layout

  // per-thread streaming pointers, stepped by BK halves each K-step
  const _Float16* gA0 = A  + (size_t)(rowBlock      + r0) * K + off0;
  const _Float16* gA1 = A  + (size_t)(rowBlock + 64 + r0) * K + off0;
  const _Float16* gB0 = Wt + (size_t)(colBlock      + r0) * K + off0;
  const _Float16* gB1 = Wt + (size_t)(colBlock + 64 + r0) * K + off0;
  _Float16* lA0[2] = { &As[0][r0 * LDSS + off0],        &As[1][r0 * LDSS + off0] };
  _Float16* lA1[2] = { &As[0][(64 + r0) * LDSS + off0], &As[1][(64 + r0) * LDSS + off0] };
  _Float16* lB0[2] = { &Bs[0][r0 * LDSS + off0],        &Bs[1][r0 * LDSS + off0] };
  _Float16* lB1[2] = { &Bs[0][(64 + r0) * LDSS + off0], &Bs[1][(64 + r0) * LDSS + off0] };

  auto prefetch = [&](int buf) {
    async_b128(gA0, lA0[buf]);
    async_b128(gA1, lA1[buf]);
    async_b128(gB0, lB0[buf]);
    async_b128(gB1, lB1[buf]);
    gA0 += BK; gA1 += BK; gB0 += BK; gB1 += BK;
  };

  auto compute = [&](int buf) {
    const _Float16* Ab = As[buf];
    const _Float16* Bb = Bs[buf];
    v16h af[2], bf[4];
#pragma unroll
    for (int tm = 0; tm < 2; ++tm) {
      // A 16x32 frag: lanes 0-15 row M=lrow K {0..7,16..23}; lanes 16-31 K {8..15,24..31}
      const _Float16* pa = Ab + (wm * 32 + tm * 16 + lrow) * LDSS + lhalf;
      union { v16h v; uint4 q[2]; } u;
      u.q[0] = *(const uint4*)pa;
      u.q[1] = *(const uint4*)(pa + 16);
      af[tm] = u.v;
    }
#pragma unroll
    for (int tn = 0; tn < 4; ++tn) {
      // B 32x16 frag: same pattern on pre-transposed Wt rows (columns of W)
      const _Float16* pb = Bb + (wn * 64 + tn * 16 + lrow) * LDSS + lhalf;
      union { v16h v; uint4 q[2]; } u;
      u.q[0] = *(const uint4*)pb;
      u.q[1] = *(const uint4*)(pb + 16);
      bf[tn] = u.v;
    }
#pragma unroll
    for (int tm = 0; tm < 2; ++tm)
#pragma unroll
      for (int tn = 0; tn < 4; ++tn)
        acc[tm][tn] = __builtin_amdgcn_wmma_f32_16x16x32_f16(
            false, af[tm], false, bf[tn], (short)0, acc[tm][tn], false, false);
  };

  const int nk = K / BK;
  prefetch(0);
  // steady state: branch-free body (prefetch next; wait stage; barrier; wmma; barrier)
  for (int i = 0; i < nk - 1; ++i) {
    const int buf = i & 1;
    prefetch(buf ^ 1);
    asm volatile("s_wait_asynccnt 0x4" ::: "memory");  // stage i landed, stage i+1 in flight
    __syncthreads();
    compute(buf);
    __syncthreads();  // all waves done reading stage i before it is overwritten
  }
  // peeled last tile
  asm volatile("s_wait_asynccnt 0x0" ::: "memory");
  __syncthreads();
  compute((nk - 1) & 1);

  // C/D layout: VGPR e holds row (e + 8*(lane>>4)), col (lane&15)
  const int rhi = (lane >> 4) * 8;
#pragma unroll
  for (int tm = 0; tm < 2; ++tm) {
#pragma unroll
    for (int tn = 0; tn < 4; ++tn) {
      const int col = colBlock + wn * 64 + tn * 16 + lrow;
      const int rbase = rowBlock + wm * 32 + tm * 16 + rhi;
      if (MODE == 0) {
        if (col < out_stride) {
          float bv = (col < n_valid) ? bias[col] : 0.f;
          size_t o = (size_t)rbase * out_stride + col;
#pragma unroll
          for (int e = 0; e < 8; ++e, o += out_stride) {
            _Float16 h = (_Float16)0.f;
            if (col < n_valid) {
              float t = acc[tm][tn][e] + bv;
              h = (_Float16)(t > 0.f ? t : 0.f);
            }
            outH[o] = h;
          }
        }
      } else if (MODE == 1) {
        if (col < n_valid) {
          float bv = bias[col];
          size_t o = (size_t)rbase * out_stride + col;
#pragma unroll
          for (int e = 0; e < 8; ++e, o += out_stride) {
            if (rbase + e < m_valid) outF[o] = acc[tm][tn][e] + bv;
          }
        }
      } else {
        size_t o = (size_t)rbase * out_stride + 12 + col;
#pragma unroll
        for (int e = 0; e < 8; ++e, o += out_stride)
          outH[o] = (_Float16)(acc[tm][tn][e] * scale[rbase + e]);
      }
    }
  }
}

// ---------------- launch ----------------

extern "C" void kernel_launch(void* const* d_in, const int* in_sizes, int n_in,
                              void* d_out, int out_size, void* d_ws, size_t ws_size,
                              hipStream_t stream) {
  const float* mixer    = (const float*)d_in[0];
  const float* features = (const float*)d_in[1];
  const int*   sb       = (const int*)d_in[2];
  const int*   mapper   = (const int*)d_in[3];
  const float* W0 = (const float*)d_in[4];
  const float* b0 = (const float*)d_in[5];
  const float* W1 = (const float*)d_in[6];
  const float* b1 = (const float*)d_in[7];
  const float* W2 = (const float*)d_in[8];
  const float* b2 = (const float*)d_in[9];
  const float* W3 = (const float*)d_in[10];
  const float* b3 = (const float*)d_in[11];
  float* out = (float*)d_out;

  const int E = in_sizes[2];
  const int n = 10000, Mreal = 40000, Mpad = 40064;  // 313 * 128

  size_t off = 0;
  auto take = [&](size_t bytes) -> char* {
    char* r = (char*)d_ws + off;
    off = (off + bytes + 255) & ~(size_t)255;
    return r;
  };
  float*    H     = (float*)take((size_t)Mpad * 128 * 4);   // (row,src) histogram
  _Float16* Hf    = (_Float16*)take((size_t)Mpad * 128 * 2);
  float*    inv   = (float*)take((size_t)Mpad * 4);         // 1/count
  _Float16* flatT = (_Float16*)take((size_t)768 * 128 * 2);
  _Float16* Wt0   = (_Float16*)take((size_t)896 * 800 * 2);
  _Float16* Wt1   = (_Float16*)take((size_t)896 * 800 * 2);
  _Float16* Wt2   = (_Float16*)take((size_t)896 * 800 * 2);
  _Float16* Wt3   = (_Float16*)take((size_t)128 * 800 * 2);
  _Float16* actA  = (_Float16*)take((size_t)Mpad * 800 * 2);
  _Float16* actB  = (_Float16*)take((size_t)Mpad * 800 * 2);
  (void)ws_size; (void)n_in; (void)out_size;

  k_zero_f32<<<(Mpad * 128 + 255) / 256, 256, 0, stream>>>(H, Mpad * 128);
  k_flatT<<<(768 * 128 + 255) / 256, 256, 0, stream>>>(mixer, flatT);
  k_hist<<<(E + 255) / 256, 256, 0, stream>>>(sb, mapper, H, E, n);
  k_finish_h<<<Mpad, 128, 0, stream>>>(H, Hf, inv, Mreal);
  k_prep_w<<<(896 * 800 + 255) / 256, 256, 0, stream>>>(W0, Wt0, 780, 780, 896, 800);
  k_prep_w<<<(896 * 800 + 255) / 256, 256, 0, stream>>>(W1, Wt1, 780, 780, 896, 800);
  k_prep_w<<<(896 * 800 + 255) / 256, 256, 0, stream>>>(W2, Wt2, 780, 780, 896, 800);
  k_prep_w<<<(128 * 800 + 255) / 256, 256, 0, stream>>>(W3, Wt3, 12, 780, 128, 800);

  dim3 blk(256);
  // nodes-mean as GEMM: actA[:,12:780) = (H @ flat) * inv
  k_gemm<2><<<dim3(Mpad / BM, 768 / BN), blk, 0, stream>>>(Hf, flatT, nullptr, inv, actA,
                                                           nullptr, 128, 768, 800, Mpad);
  k_edges<<<Mpad, 32, 0, stream>>>(features, actA, Mreal);
  // 4-layer MLP
  k_gemm<0><<<dim3(Mpad / BM, 896 / BN), blk, 0, stream>>>(actA, Wt0, b0, nullptr, actB,
                                                           nullptr, 800, 780, 800, Mpad);
  k_gemm<0><<<dim3(Mpad / BM, 896 / BN), blk, 0, stream>>>(actB, Wt1, b1, nullptr, actA,
                                                           nullptr, 800, 780, 800, Mpad);
  k_gemm<0><<<dim3(Mpad / BM, 896 / BN), blk, 0, stream>>>(actA, Wt2, b2, nullptr, actB,
                                                           nullptr, 800, 780, 800, Mpad);
  k_gemm<1><<<dim3(Mpad / BM, 128 / BN), blk, 0, stream>>>(actB, Wt3, b3, nullptr, nullptr,
                                                           out, 800, 12, 12, Mreal);
}
